// SimpleSNNModel2D_79216376807541
// MI455X (gfx1250) — matrix operand order repaired and checked by reading, not compiled, not run on hardware
//
#include <hip/hip_runtime.h>

typedef _Float16 v16h __attribute__((ext_vector_type(16)));
typedef _Float16 v8h  __attribute__((ext_vector_type(8)));
typedef float    v8f  __attribute__((ext_vector_type(8)));

#define HCH   128      // hidden channels
#define NP    64       // positions per workgroup (4 t-rows x 16 v)
#define TSTEPS 4
#define VTHR  0.5f
#define EPSV  1e-5f

// ---------------------------------------------------------------------------
// Main fused SNN kernel: fc1+BN1 -> S x (LIF1 -> WMMA fc2 -> BN2 -> LIF2)
// -> per-(b,channel) spike-count accumulation via atomics into featAcc.
// ---------------------------------------------------------------------------
__global__ __launch_bounds__(256) void snn_main(
    const float* __restrict__ x,
    const float* __restrict__ w1, const float* __restrict__ b1,
    const float* __restrict__ g1, const float* __restrict__ be1,
    const float* __restrict__ m1, const float* __restrict__ rv1,
    const float* __restrict__ w2, const float* __restrict__ b2,
    const float* __restrict__ g2, const float* __restrict__ be2,
    const float* __restrict__ m2, const float* __restrict__ rv2,
    float* __restrict__ featAcc)
{
    __shared__ __attribute__((aligned(32))) _Float16 lds_w2[HCH * HCH]; // [m][k] f16
    __shared__ __attribute__((aligned(32))) _Float16 lds_s1[NP * HCH];  // [pos][ch] f16
    __shared__ float lds_bn[2 * HCH];                                   // inv2, off2

    const int tid  = threadIdx.x;
    const int lane = tid & 31;
    const int wv   = tid >> 5;      // wave 0..7 -> owns channel tile [16wv,16wv+16)
    const int half = lane >> 4;     // lane half selects K sub-range in fragments
    const int l16  = lane & 15;     // M (A) / N (B,C,D) index within tile

    const int b    = blockIdx.x >> 7;   // 128 tiles per batch element
    const int tile = blockIdx.x & 127;
    const int t0   = tile * 4;          // 4 time rows x 16 vertices = 64 positions

    // --- preload w2 as f16 into LDS (A-matrix source, reused S*4*4 times) ---
    for (int i = tid; i < HCH * HCH; i += 256) lds_w2[i] = (_Float16)w2[i];

    // --- folded BN2 constants ---
    if (tid < HCH) {
        float inv2 = g2[tid] * rsqrtf(rv2[tid] + EPSV);
        lds_bn[tid]       = inv2;
        lds_bn[HCH + tid] = (b2[tid] - m2[tid]) * inv2 + be2[tid];
    }

    // --- stage 1: fc1 + BN1, thread owns channel (tid&127), 32 positions ---
    const int ch      = tid & 127;
    const int posBase = (tid >> 7) * 32;
    float inv1 = g1[ch] * rsqrtf(rv1[ch] + EPSV);
    float off1 = (b1[ch] - m1[ch]) * inv1 + be1[ch];
    float w1c0 = w1[ch * 3 + 0], w1c1 = w1[ch * 3 + 1], w1c2 = w1[ch * 3 + 2];

    float h1r[32], v1r[32];
    #pragma unroll
    for (int i = 0; i < 32; ++i) {
        int pos = posBase + i;
        int t   = t0 + (pos >> 4);
        int v   = pos & 15;
        const float* xp = x + (((size_t)b * 512 + t) * 3) * 16 + v; // x[b][t][c][v]
        float conv = w1c0 * xp[0] + w1c1 * xp[16] + w1c2 * xp[32];
        h1r[i] = conv * inv1 + off1;
        v1r[i] = 0.0f;
    }

    __syncthreads();

    // --- stage 2 persistent WMMA A fragments: w2 rows [16wv,16wv+16) ---
    // 16-bit A layout: lane l16 = M; half selects K 16..31 (contiguous).
    v16h afrag[4];
    #pragma unroll
    for (int kc = 0; kc < 4; ++kc) {
        const _Float16* rowp = lds_w2 + (16 * wv + l16) * HCH + 32 * kc + 16 * half;
        v8h lo = *(const v8h*)(rowp);        // K = 32kc + 16h + [0..7]
        v8h hi = *(const v8h*)(rowp + 8);    // K = 32kc + 16h + 8 + [0..7]
        v16h a;
        #pragma unroll
        for (int e = 0; e < 8; ++e) { a[e] = lo[e]; a[8 + e] = hi[e]; }
        afrag[kc] = a;
    }

    // per-lane BN2 constants for the 8 accumulator rows (M = 16wv + j + 8*half)
    float inv2v[8], off2v[8];
    #pragma unroll
    for (int j = 0; j < 8; ++j) {
        int m    = 16 * wv + j + 8 * half;
        inv2v[j] = lds_bn[m];
        off2v[j] = lds_bn[HCH + m];
    }

    v8f v2t[4], acct[4];
    #pragma unroll
    for (int nt = 0; nt < 4; ++nt) { v2t[nt] = {}; acct[nt] = {}; }

    for (int s = 0; s < TSTEPS; ++s) {
        // LIF1 (decay_input, hard reset) + spike write to LDS [pos][ch] as f16
        #pragma unroll
        for (int i = 0; i < 32; ++i) {
            float v1n = 0.5f * (v1r[i] + h1r[i]);
            float sp  = (v1n >= VTHR) ? 1.0f : 0.0f;
            v1r[i] = v1n * (1.0f - sp);
            lds_s1[(posBase + i) * HCH + ch] = (_Float16)sp;
        }
        __syncthreads();

        // fc2: wave wv computes channels [16wv,16wv+16) x 64 positions
        #pragma unroll
        for (int nt = 0; nt < 4; ++nt) {
            v8f c = {};
            #pragma unroll
            for (int kc = 0; kc < 4; ++kc) {
                // B layout: lane l16 = N, half selects K 16..31; contiguous 32B
                const _Float16* bp =
                    lds_s1 + (16 * nt + l16) * HCH + 32 * kc + 16 * half;
                v16h bfrag = *(const v16h*)bp;
                c = __builtin_amdgcn_wmma_f32_16x16x32_f16(
                        false, afrag[kc], false, bfrag, (short)0, c, false, false);
            }
            // BN2 + LIF2 + spike accumulation, all in registers
            #pragma unroll
            for (int j = 0; j < 8; ++j) {
                float h2  = c[j] * inv2v[j] + off2v[j];
                float v2n = 0.5f * (v2t[nt][j] + h2);
                float sp  = (v2n >= VTHR) ? 1.0f : 0.0f;
                acct[nt][j] += sp;
                v2t[nt][j] = v2n * (1.0f - sp);
            }
        }
        __syncthreads();
    }

    // --- reduce spike counts over the 64 positions of this tile ---
    // C/D layout: lane l16 = N, so sum the 4 N-tiles then fold 16 lanes/half.
    #pragma unroll
    for (int j = 0; j < 8; ++j) {
        float a = acct[0][j] + acct[1][j] + acct[2][j] + acct[3][j];
        a += __shfl_xor(a, 1, 32);
        a += __shfl_xor(a, 2, 32);
        a += __shfl_xor(a, 4, 32);
        a += __shfl_xor(a, 8, 32);
        if (l16 == 0) {
            int m = 16 * wv + j + 8 * half;
            atomicAdd(&featAcc[b * HCH + m], a);
        }
    }
}

// ---------------------------------------------------------------------------
__global__ void zero_ws(float* __restrict__ p, int n) {
    int i = blockIdx.x * 256 + threadIdx.x;
    if (i < n) p[i] = 0.0f;
}

// head: out[b][nc] = bc[nc] + sum_h featAcc[b][h] * wc[nc][h] / (S*T*V)
__global__ __launch_bounds__(256) void snn_head(
    const float* __restrict__ featAcc,
    const float* __restrict__ wc, const float* __restrict__ bc,
    float* __restrict__ out)
{
    int tid = blockIdx.x * blockDim.x + threadIdx.x;
    if (tid >= 16 * 12) return;
    int bb = tid / 12, nc = tid % 12;
    float s = 0.0f;
    #pragma unroll 4
    for (int h = 0; h < HCH; ++h)
        s += featAcc[bb * HCH + h] * wc[nc * HCH + h];
    out[tid] = bc[nc] + s * (1.0f / 32768.0f);   // 1 / (S * T * V)
}

// ---------------------------------------------------------------------------
extern "C" void kernel_launch(void* const* d_in, const int* in_sizes, int n_in,
                              void* d_out, int out_size, void* d_ws, size_t ws_size,
                              hipStream_t stream) {
    const float* x   = (const float*)d_in[0];
    const float* w1  = (const float*)d_in[1];
    const float* b1  = (const float*)d_in[2];
    const float* g1  = (const float*)d_in[3];
    const float* be1 = (const float*)d_in[4];
    const float* m1  = (const float*)d_in[5];
    const float* rv1 = (const float*)d_in[6];
    const float* w2  = (const float*)d_in[7];
    const float* b2  = (const float*)d_in[8];
    const float* g2  = (const float*)d_in[9];
    const float* be2 = (const float*)d_in[10];
    const float* m2  = (const float*)d_in[11];
    const float* rv2 = (const float*)d_in[12];
    const float* wc  = (const float*)d_in[13];
    const float* bc  = (const float*)d_in[14];

    float* featAcc = (float*)d_ws;          // 16 * 128 floats
    const int nAcc = 16 * HCH;

    zero_ws<<<(nAcc + 255) / 256, 256, 0, stream>>>(featAcc, nAcc);

    // 16 batches * (8192 positions / 64 per WG) = 2048 workgroups
    snn_main<<<2048, 256, 0, stream>>>(x, w1, b1, g1, be1, m1, rv1,
                                       w2, b2, g2, be2, m2, rv2, featAcc);

    snn_head<<<1, 256, 0, stream>>>(featAcc, wc, bc, (float*)d_out);
}
